// DDIGNN_66305705116452
// MI455X (gfx1250) — matrix operand order repaired and checked by reading, not compiled, not run on hardware
//
#include <hip/hip_runtime.h>
#include <math.h>

#define Nn   100000
#define Ee   1600000
#define Pp   500000
#define Hc   128
#define NHc  (Nn * Hc)
#define WFRAG_ELEMS (4 * 8 * 32 * 16)   // kchunks * ntiles * lanes * elems = 16384 halfs = 32KB

typedef __attribute__((ext_vector_type(16))) _Float16 v16h;
typedef __attribute__((ext_vector_type(8)))  float    v8f;

// ---------------- degree / dinv ----------------
__global__ void k_fill1(float* p, int n) {
    int idx = blockIdx.x * blockDim.x + threadIdx.x;
    if (idx < n) p[idx] = 1.0f;
}

__global__ void k_count_deg(const int* __restrict__ dst, float* deg, int e) {
    int idx = blockIdx.x * blockDim.x + threadIdx.x;
    if (idx < e) atomicAdd(&deg[dst[idx]], 1.0f);
}

__global__ void k_rsqrt(float* p, int n) {
    int idx = blockIdx.x * blockDim.x + threadIdx.x;
    if (idx < n) p[idx] = rsqrtf(p[idx]);
}

// ---------------- W repack: f32 [128x128] row-major -> f16 B-fragment layout ----------------
// Fragment (kcI, t), lane, elem e holds W[(kcI*32 + 8*(lane>=16) + (e&7) + 16*(e>=8))*128 + t*16 + (lane&15)]
// stored contiguously so the GEMM loads each v16h fragment as one 32B vector load.
__global__ void k_packW(const float* __restrict__ W, _Float16* __restrict__ Wf) {
    int idx = blockIdx.x * blockDim.x + threadIdx.x;
    if (idx >= WFRAG_ELEMS) return;
    const int e    = idx & 15;
    const int lane = (idx >> 4) & 31;
    const int frag = idx >> 9;
    const int t    = frag & 7;
    const int kcI  = frag >> 3;
    const int K    = kcI * 32 + ((lane >> 4) << 3) + (e & 7) + ((e >> 3) << 4);
    const int col  = t * 16 + (lane & 15);
    Wf[idx] = (_Float16)W[(size_t)K * Hc + col];
}

// ---------------- WMMA GEMM: Out[n][c] = sum_k A[n][k] * W[k][c] ----------------
// A: [nrows x 128] f32 row-major (converted to f16 in registers),
// Wf: fragment-packed f16 weights, Out: [nrows x 128] f32.
// 4 waves/block, one 16-row tile per wave, f32 accumulation via v_wmma_f32_16x16x32_f16.
__global__ __launch_bounds__(128) void k_gemm_wmma(const float* __restrict__ A,
                                                   const _Float16* __restrict__ Wf,
                                                   float* __restrict__ Out,
                                                   int nrows) {
    const int lane = threadIdx.x & 31;
    const int wave = threadIdx.x >> 5;
    const int row0 = (blockIdx.x * 4 + wave) * 16;
    if (row0 >= nrows) return;                 // wave-uniform branch: EXEC stays all-ones

    const int half = lane >> 4;                // 0: lanes 0-15, 1: lanes 16-31
    const int mrow = lane & 15;                // M index within tile / N index for B & C
    const int row  = row0 + mrow;
    const bool rv  = (row < nrows);
    const float* arow = A + (size_t)(rv ? row : row0) * Hc;  // clamp: garbage rows never stored

    v8f acc[8] = {};

    #pragma unroll
    for (int kcI = 0; kcI < 4; ++kcI) {
        const int kc   = kcI * 32;
        const int koff = half * 8;
        // A fragment: 16 halfs = K {kc+koff+0..7, kc+16+koff+0..7} of row `row`
        const float4* p0 = reinterpret_cast<const float4*>(arow + kc + koff);
        const float4* p1 = reinterpret_cast<const float4*>(arow + kc + 16 + koff);
        float4 f0 = p0[0], f1 = p0[1];
        float4 f2 = p1[0], f3 = p1[1];
        v16h a;
        a[0]=(_Float16)f0.x; a[1]=(_Float16)f0.y; a[2]=(_Float16)f0.z; a[3]=(_Float16)f0.w;
        a[4]=(_Float16)f1.x; a[5]=(_Float16)f1.y; a[6]=(_Float16)f1.z; a[7]=(_Float16)f1.w;
        a[8]=(_Float16)f2.x; a[9]=(_Float16)f2.y; a[10]=(_Float16)f2.z; a[11]=(_Float16)f2.w;
        a[12]=(_Float16)f3.x; a[13]=(_Float16)f3.y; a[14]=(_Float16)f3.z; a[15]=(_Float16)f3.w;

        #pragma unroll
        for (int t = 0; t < 8; ++t) {
            // one contiguous 32B fragment load per tile (was 16 strided b32 + cvts)
            const v16h b = *reinterpret_cast<const v16h*>(
                Wf + (((size_t)(kcI * 8 + t) * 32 + lane) << 4));
            acc[t] = __builtin_amdgcn_wmma_f32_16x16x32_f16(
                /*neg_a=*/false, a, /*neg_b=*/false, b,
                /*c_mod=*/(short)0, acc[t], /*reuse_a=*/false, /*reuse_b=*/false);
        }
    }

    // C layout: VGPR r -> row (r + 8*half), col = mrow
    #pragma unroll
    for (int t = 0; t < 8; ++t) {
        const int n0 = t * 16;
        #pragma unroll
        for (int r = 0; r < 8; ++r) {
            const int orow = row0 + r + half * 8;
            if (orow < nrows) Out[(size_t)orow * Hc + n0 + mrow] = acc[t][r];
        }
    }
}

// ---------------- agg init: agg = dinv^2 * h (self-loop term) ----------------
__global__ void k_selfloop(const float* __restrict__ h, const float* __restrict__ dinv,
                           float* __restrict__ agg, int n) {
    int idx = blockIdx.x * blockDim.x + threadIdx.x;
    if (idx < n) {
        float dv = dinv[idx >> 7];
        agg[idx] = dv * dv * h[idx];
    }
}

// ---------------- edge scatter: agg[dst] += h[src] * dinv[src]*dinv[dst] ----------------
// one block per edge, one lane per channel; gathers + atomics hit L2 (h = 51MB << 192MB L2)
__global__ __launch_bounds__(128) void k_scatter(const float* __restrict__ h,
                                                 const int* __restrict__ src,
                                                 const int* __restrict__ dst,
                                                 const float* __restrict__ dinv,
                                                 float* __restrict__ agg) {
    const int e = blockIdx.x;
    const int c = threadIdx.x;
    const int s = src[e], d = dst[e];
    const float nrm = dinv[s] * dinv[d];
    atomicAdd(&agg[(size_t)d * Hc + c], h[(size_t)s * Hc + c] * nrm);
}

// ---------------- bias (+optional relu), in-place ----------------
__global__ void k_bias_act(float* __restrict__ agg, const float* __restrict__ bias,
                           int n, int do_relu) {
    int idx = blockIdx.x * blockDim.x + threadIdx.x;
    if (idx < n) {
        float v = agg[idx] + bias[idx & (Hc - 1)];
        if (do_relu) v = fmaxf(v, 0.0f);
        agg[idx] = v;
    }
}

// ---------------- link predictor: sigmoid([h[i], h[j]] @ Wlin + b) ----------------
// one wave32 per pair; float4 loads; wave shuffle reduction
__global__ __launch_bounds__(256) void k_predict(const float* __restrict__ h,
                                                 const int* __restrict__ pi,
                                                 const int* __restrict__ pj,
                                                 const float* __restrict__ wlin,
                                                 const float* __restrict__ blin,
                                                 float* __restrict__ out, int np) {
    const int p = blockIdx.x * 8 + (threadIdx.x >> 5);
    if (p >= np) return;
    const int lane = threadIdx.x & 31;
    const int ni = pi[p], nj = pj[p];
    const float4 xi = reinterpret_cast<const float4*>(h + (size_t)ni * Hc)[lane];
    const float4 xj = reinterpret_cast<const float4*>(h + (size_t)nj * Hc)[lane];
    const float4 wi = reinterpret_cast<const float4*>(wlin)[lane];
    const float4 wj = reinterpret_cast<const float4*>(wlin + Hc)[lane];
    float sum = xi.x * wi.x + xi.y * wi.y + xi.z * wi.z + xi.w * wi.w
              + xj.x * wj.x + xj.y * wj.y + xj.z * wj.z + xj.w * wj.w;
    #pragma unroll
    for (int off = 16; off > 0; off >>= 1) sum += __shfl_down(sum, off, 32);
    if (lane == 0) {
        float z = sum + blin[0];
        out[p] = 1.0f / (1.0f + __expf(-z));
    }
}

// ---------------- launch ----------------
extern "C" void kernel_launch(void* const* d_in, const int* in_sizes, int n_in,
                              void* d_out, int out_size, void* d_ws, size_t ws_size,
                              hipStream_t stream) {
    (void)in_sizes; (void)n_in; (void)out_size; (void)ws_size;
    const float* x    = (const float*)d_in[0];
    const int*   eix  = (const int*)d_in[1];     // [2, E] row-major
    const int*   pi   = (const int*)d_in[2];
    const int*   pj   = (const int*)d_in[3];
    const float* W1   = (const float*)d_in[4];
    const float* b1   = (const float*)d_in[5];
    const float* W2   = (const float*)d_in[6];
    const float* b2   = (const float*)d_in[7];
    const float* Wlin = (const float*)d_in[8];
    const float* blin = (const float*)d_in[9];
    float* out = (float*)d_out;

    const int* src = eix;
    const int* dst = eix + Ee;

    float*     buf0 = (float*)d_ws;                    // h
    float*     buf1 = buf0 + (size_t)NHc;              // agg / layer output
    float*     dinv = buf1 + (size_t)NHc;              // deg -> dinv
    _Float16*  Wf1  = (_Float16*)(dinv + Nn);          // 32B-aligned (offset 102,800,000 bytes)
    _Float16*  Wf2  = Wf1 + WFRAG_ELEMS;

    const int TB = 256;
    const int gN  = (Nn + TB - 1) / TB;
    const int gE  = (Ee + TB - 1) / TB;
    const int gNH = (NHc + TB - 1) / TB;
    const int gW  = (WFRAG_ELEMS + TB - 1) / TB;
    const int gGemm = (Nn + 63) / 64;      // 4 waves/block * 16 rows/wave

    // one-time weight repacks into WMMA B-fragment layout
    k_packW<<<gW, TB, 0, stream>>>(W1, Wf1);
    k_packW<<<gW, TB, 0, stream>>>(W2, Wf2);

    // degrees with self loops -> dinv
    k_fill1<<<gN, TB, 0, stream>>>(dinv, Nn);
    k_count_deg<<<gE, TB, 0, stream>>>(dst, dinv, Ee);
    k_rsqrt<<<gN, TB, 0, stream>>>(dinv, Nn);

    // layer 1: h1 = X @ W1 ; agg = D^-1/2 (A+I) D^-1/2 h1 + b1 ; relu
    k_gemm_wmma<<<gGemm, 128, 0, stream>>>(x, Wf1, buf0, Nn);
    k_selfloop<<<gNH, TB, 0, stream>>>(buf0, dinv, buf1, NHc);
    k_scatter<<<Ee, 128, 0, stream>>>(buf0, src, dst, dinv, buf1);
    k_bias_act<<<gNH, TB, 0, stream>>>(buf1, b1, NHc, 1);

    // layer 2 (no relu)
    k_gemm_wmma<<<gGemm, 128, 0, stream>>>(buf1, Wf2, buf0, Nn);
    k_selfloop<<<gNH, TB, 0, stream>>>(buf0, dinv, buf1, NHc);
    k_scatter<<<Ee, 128, 0, stream>>>(buf0, src, dst, dinv, buf1);
    k_bias_act<<<gNH, TB, 0, stream>>>(buf1, b2, NHc, 0);

    // link prediction
    k_predict<<<(Pp + 7) / 8, 256, 0, stream>>>(buf1, pi, pj, Wlin, blin, out, Pp);
}
